// ArcLoss_5823975653775
// MI455X (gfx1250) — matrix operand order repaired and checked by reading, not compiled, not run on hardware
//
#include <hip/hip_runtime.h>
#include <hip/hip_bf16.h>

typedef __attribute__((ext_vector_type(16))) __bf16 v16bf;
typedef __attribute__((ext_vector_type(8)))  __bf16 v8bf;
typedef __attribute__((ext_vector_type(4)))  __bf16 v4bf;
typedef __attribute__((ext_vector_type(8)))  float  v8f;

#define S_SCALE 30.0f
#define MARGIN  0.4f
#define B_ROWS  1024
#define D_DIM   512
#define C_CLS   50000
#define NBLKN   196   /* ceil(50000 / 256) */

// ---- workspace layout (bytes) ----
#define WS_PART_OFF  0u
#define WS_PART_SZ   (B_ROWS * NBLKN * 4u)                  /* 802816  */
#define WS_LOSS_OFF  (WS_PART_OFF + WS_PART_SZ)
#define WS_LOSS_SZ   (B_ROWS * 4u)
#define WS_XH_OFF    (WS_LOSS_OFF + WS_LOSS_SZ)
#define WS_X_SZ      (B_ROWS * D_DIM * 2u)                  /* 1 MB    */
#define WS_XL_OFF    (WS_XH_OFF + WS_X_SZ)
#define WS_WH_OFF    (WS_XL_OFF + WS_X_SZ)                  /* 64B-aligned */
#define WS_W_SZ      ((size_t)C_CLS * D_DIM * 2u)           /* 51.2 MB */
#define WS_WL_OFF    (WS_WH_OFF + WS_W_SZ)
#define WS_NEED_PRE  (WS_WL_OFF + WS_W_SZ)                  /* ~105.3 MB */
#define WS_NEED_MIN  WS_WH_OFF                              /* ~2.9 MB  */

static __device__ __forceinline__ float4 ld4(const float* p) {
    return *reinterpret_cast<const float4*>(p);
}
static __device__ __forceinline__ v8bf ldb8(const __bf16* p) {
    return *reinterpret_cast<const v8bf*>(p);
}
static __device__ __forceinline__ v16bf cat8(v8bf a, v8bf b) {
    return __builtin_shufflevector(a, b, 0,1,2,3,4,5,6,7,8,9,10,11,12,13,14,15);
}
static __device__ __forceinline__ void split1(float f, __bf16& hi, __bf16& lo) {
    hi = (__bf16)f;                 // RNE
    lo = (__bf16)(f - (float)hi);   // residual
}
static __device__ __forceinline__ void split16(const float* t, v16bf& hi, v16bf& lo) {
#pragma unroll
    for (int e = 0; e < 16; ++e) { __bf16 h, l; split1(t[e], h, l); hi[e] = h; lo[e] = l; }
}

// ---------------- Phase 1: row norms of x + normalized bf16 hi/lo split ----------------
__global__ __launch_bounds__(256)
void row_prep(const float* __restrict__ x, __bf16* __restrict__ xh, __bf16* __restrict__ xl)
{
    __shared__ float sm[256];
    const int row = blockIdx.x;
    const int tid = threadIdx.x;
    float ss = 0.0f;
    float v0 = x[(size_t)row * D_DIM + tid];
    float v1 = x[(size_t)row * D_DIM + tid + 256];
    ss = v0 * v0 + v1 * v1;
    sm[tid] = ss;
    __syncthreads();
    for (int s = 128; s > 0; s >>= 1) {
        if (tid < s) sm[tid] += sm[tid + s];
        __syncthreads();
    }
    const float inv = 1.0f / fmaxf(sqrtf(sm[0]), 1e-12f);
    __bf16 h, l;
    split1(v0 * inv, h, l);
    xh[(size_t)row * D_DIM + tid] = h;  xl[(size_t)row * D_DIM + tid] = l;
    split1(v1 * inv, h, l);
    xh[(size_t)row * D_DIM + tid + 256] = h;  xl[(size_t)row * D_DIM + tid + 256] = l;
}

// ---------------- Phase 1b: one-time bf16 hi/lo split of W ----------------
__global__ __launch_bounds__(256)
void w_split(const float* __restrict__ W, __bf16* __restrict__ Wh, __bf16* __restrict__ Wl)
{
    const size_t i4 = ((size_t)blockIdx.x * 256 + threadIdx.x) * 4;  // 4 elements/thread
    float4 f = ld4(W + i4);
    v4bf h, l;
    {
        __bf16 hh, ll;
        split1(f.x, hh, ll); h[0] = hh; l[0] = ll;
        split1(f.y, hh, ll); h[1] = hh; l[1] = ll;
        split1(f.z, hh, ll); h[2] = hh; l[2] = ll;
        split1(f.w, hh, ll); h[3] = hh; l[3] = ll;
    }
    *reinterpret_cast<v4bf*>(Wh + i4) = h;
    *reinterpret_cast<v4bf*>(Wl + i4) = l;
}

// ---------------- Phase 2: bf16x3 WMMA GEMM + deterministic exp partials ----------------
// grid (1024/64, 196); 256 threads = 8 waves as 2(M) x 4(N); wave tile 32x64 -> block 64x256
template <bool PRE>
__global__ __launch_bounds__(256)
void arc_gemm(const __bf16* __restrict__ xh, const __bf16* __restrict__ xl,
              const float* __restrict__ W,
              const __bf16* __restrict__ Wh, const __bf16* __restrict__ Wl,
              float* __restrict__ cosOut, float* __restrict__ partials)
{
    __shared__ float red[64][4];

    const int tid  = threadIdx.x;
    const int wave = tid >> 5;
    const int lane = tid & 31;
    const int wm   = wave >> 2;   // 0..1
    const int wn   = wave & 3;    // 0..3
    const int m16  = lane & 15;
    const int h    = lane >> 4;

    const int rowBase = blockIdx.x * 64;
    const int colBase = blockIdx.y * 256;
    const int waveRow = rowBase + wm * 32;
    const int waveCol = colBase + wn * 64;

    v8f acc[2][4] = {};

    // A: lane m16 holds row (waveRow + mt*16 + m16); chunks at K = 16h and 16h+8
    const __bf16* pAh[2];
    const __bf16* pAl[2];
#pragma unroll
    for (int mt = 0; mt < 2; ++mt) {
        const size_t r = waveRow + mt * 16 + m16;
        pAh[mt] = xh + r * D_DIM;
        pAl[mt] = xl + r * D_DIM;
    }
    // B: lane m16 holds column (waveCol + nt*16 + m16) = W row; chunk at K = 16h..16h+15
    // out-of-range columns are CLAMPED (valid memory, results discarded) so that
    // every WMMA executes with full EXEC -> no exec-masked WMMA anywhere.
    int  bCol[4];
    const __bf16* pBh[4]; const __bf16* pBl[4]; const float* pBf[4];
#pragma unroll
    for (int nt = 0; nt < 4; ++nt) {
        bCol[nt] = waveCol + nt * 16 + m16;
        const size_t c = (size_t)min(bCol[nt], C_CLS - 1);
        if (PRE) { pBh[nt] = Wh + c * D_DIM; pBl[nt] = Wl + c * D_DIM; }
        else     { pBf[nt] = W  + c * D_DIM; }
    }

    for (int k = 0; k < D_DIM; k += 32) {
        v16bf ah[2], al[2], bh[4], bl[4];
#pragma unroll
        for (int mt = 0; mt < 2; ++mt) {
            const int o0 = k + 16 * h, o1 = k + 16 * h + 8;
            ah[mt] = cat8(ldb8(pAh[mt] + o0), ldb8(pAh[mt] + o1));
            al[mt] = cat8(ldb8(pAl[mt] + o0), ldb8(pAl[mt] + o1));
        }
#pragma unroll
        for (int nt = 0; nt < 4; ++nt) {
            const int o = k + 16 * h;
            if (PRE) {
                bh[nt] = cat8(ldb8(pBh[nt] + o), ldb8(pBh[nt] + o + 8));
                bl[nt] = cat8(ldb8(pBl[nt] + o), ldb8(pBl[nt] + o + 8));
            } else {
                float t[16];
                float4 c0 = ld4(pBf[nt] + o);     float4 c1 = ld4(pBf[nt] + o + 4);
                float4 c2 = ld4(pBf[nt] + o + 8); float4 c3 = ld4(pBf[nt] + o + 12);
                t[0]=c0.x;  t[1]=c0.y;  t[2]=c0.z;  t[3]=c0.w;
                t[4]=c1.x;  t[5]=c1.y;  t[6]=c1.z;  t[7]=c1.w;
                t[8]=c2.x;  t[9]=c2.y;  t[10]=c2.z; t[11]=c2.w;
                t[12]=c3.x; t[13]=c3.y; t[14]=c3.z; t[15]=c3.w;
                split16(t, bh[nt], bl[nt]);
            }
        }
#pragma unroll
        for (int mt = 0; mt < 2; ++mt) {
#pragma unroll
            for (int nt = 0; nt < 4; ++nt) {
                acc[mt][nt] = __builtin_amdgcn_wmma_f32_16x16x32_bf16(
                    false, ah[mt], false, bh[nt], (short)0, acc[mt][nt], false, false);
                acc[mt][nt] = __builtin_amdgcn_wmma_f32_16x16x32_bf16(
                    false, ah[mt], false, bl[nt], (short)0, acc[mt][nt], false, false);
                acc[mt][nt] = __builtin_amdgcn_wmma_f32_16x16x32_bf16(
                    false, al[mt], false, bh[nt], (short)0, acc[mt][nt], false, false);
            }
        }
    }

    // ---- store cos_theta (non-temporal: keep W resident in the 192MB L2) ----
    // D layout: VGPR r, lanes 0-15 -> (M=r, N=lane); lanes 16-31 -> (M=r+8, N=lane-16)
#pragma unroll
    for (int mt = 0; mt < 2; ++mt) {
#pragma unroll
        for (int nt = 0; nt < 4; ++nt) {
            if (bCol[nt] < C_CLS) {   // per-lane predicate on stores only
                const size_t base = (size_t)(waveRow + mt * 16 + 8 * h) * C_CLS + bCol[nt];
#pragma unroll
                for (int r = 0; r < 8; ++r)
                    __builtin_nontemporal_store(acc[mt][nt][r],
                                                cosOut + base + (size_t)r * C_CLS);
            }
        }
    }

    // ---- deterministic per-row sum of exp(S*cos) over this block's columns ----
#pragma unroll
    for (int mt = 0; mt < 2; ++mt) {
#pragma unroll
        for (int r = 0; r < 8; ++r) {
            float v = 0.0f;
#pragma unroll
            for (int nt = 0; nt < 4; ++nt)
                v += (bCol[nt] < C_CLS) ? __expf(S_SCALE * acc[mt][nt][r]) : 0.0f;
#pragma unroll
            for (int off = 1; off < 16; off <<= 1)   // reduce each 16-lane half
                v += __shfl_xor(v, off, 32);
            if (m16 == 0)   // lane 0 -> row r, lane 16 -> row r+8
                red[wm * 32 + mt * 16 + r + 8 * h][wn] = v;
        }
    }
    __syncthreads();
    if (tid < 64) {
        const float p = red[tid][0] + red[tid][1] + red[tid][2] + red[tid][3];
        partials[(size_t)(rowBase + tid) * NBLKN + blockIdx.y] = p;
    }
}

// ---------------- Phase 3: per-row ArcFace loss ----------------
__global__ __launch_bounds__(128)
void row_loss(const float* __restrict__ partials, const float* __restrict__ cosOut,
              const int* __restrict__ labels, float* __restrict__ lossbuf)
{
    __shared__ float sm[128];
    const int row = blockIdx.x;
    float a = 0.0f;
    for (int t = threadIdx.x; t < NBLKN; t += 128)
        a += partials[(size_t)row * NBLKN + t];
    sm[threadIdx.x] = a;
    __syncthreads();
    for (int s = 64; s > 0; s >>= 1) {
        if (threadIdx.x < s) sm[threadIdx.x] += sm[threadIdx.x + s];
        __syncthreads();
    }
    if (threadIdx.x == 0) {
        const float total = sm[0];
        const int   lbl   = labels[row];
        float t = cosOut[(size_t)row * C_CLS + lbl];
        t = fminf(fmaxf(t, -1.0f + 1e-7f), 1.0f - 1e-7f);
        // S*cos(acos(t)+M) = S*(t*cosM - sqrt(1-t^2)*sinM)
        const float sinT  = sqrtf(fmaxf(1.0f - t * t, 0.0f));
        const float num   = S_SCALE * (t * cosf(MARGIN) - sinT * sinf(MARGIN));
        const float excl  = total - expf(S_SCALE * t);
        const float denom = expf(num) + excl;
        lossbuf[row] = num - logf(denom);
    }
}

// ---------------- Phase 4: -mean(loss) ----------------
__global__ __launch_bounds__(256)
void mean_loss(const float* __restrict__ lossbuf, float* __restrict__ outLoss)
{
    __shared__ float sm[256];
    float a = 0.0f;
    for (int t = threadIdx.x; t < B_ROWS; t += 256)
        a += lossbuf[t];
    sm[threadIdx.x] = a;
    __syncthreads();
    for (int s = 128; s > 0; s >>= 1) {
        if (threadIdx.x < s) sm[threadIdx.x] += sm[threadIdx.x + s];
        __syncthreads();
    }
    if (threadIdx.x == 0)
        outLoss[0] = -(sm[0] / (float)B_ROWS);
}

extern "C" void kernel_launch(void* const* d_in, const int* in_sizes, int n_in,
                              void* d_out, int out_size, void* d_ws, size_t ws_size,
                              hipStream_t stream)
{
    const float* x      = (const float*)d_in[0];
    const int*   labels = (const int*)d_in[1];
    const float* W      = (const float*)d_in[2];

    float* cosOut = (float*)d_out;
    float* loss   = cosOut + (size_t)B_ROWS * C_CLS;

    char* ws = (char*)d_ws;
    float*  partials = (float*)(ws + WS_PART_OFF);
    float*  lossbuf  = (float*)(ws + WS_LOSS_OFF);
    __bf16* xh       = (__bf16*)(ws + WS_XH_OFF);
    __bf16* xl       = (__bf16*)(ws + WS_XL_OFF);
    __bf16* Wh       = (__bf16*)(ws + WS_WH_OFF);
    __bf16* Wl       = (__bf16*)(ws + WS_WL_OFF);

    const bool pre = (ws_size >= (size_t)WS_NEED_PRE);   // ws_size is call-invariant

    row_prep<<<B_ROWS, 256, 0, stream>>>(x, xh, xl);

    dim3 g2(B_ROWS / 64, NBLKN);
    if (pre) {
        w_split<<<(C_CLS * D_DIM) / (256 * 4), 256, 0, stream>>>(W, Wh, Wl);
        arc_gemm<true><<<g2, 256, 0, stream>>>(xh, xl, W, Wh, Wl, cosOut, partials);
    } else {
        arc_gemm<false><<<g2, 256, 0, stream>>>(xh, xl, W, Wh, Wl, cosOut, partials);
    }

    row_loss<<<B_ROWS, 128, 0, stream>>>(partials, cosOut, labels, lossbuf);
    mean_loss<<<1, 256, 0, stream>>>(lossbuf, loss);
}
